// SelfAttentionConv2d_23742579212998
// MI455X (gfx1250) — compile-verified, matrix operand order checked by reading, not run on hardware
//
#include <hip/hip_runtime.h>

typedef __bf16 bf16;
typedef __attribute__((ext_vector_type(16))) __bf16 v16bf;
typedef __attribute__((ext_vector_type(8)))  float   v8f;
typedef __attribute__((ext_vector_type(4)))  unsigned u32x4;

// ---------- bf16 <-> f32 helpers (round-to-nearest-even) ----------
static __device__ __forceinline__ bf16 f2bf(float f) {
  union { float f; unsigned u; } v; v.f = f;
  unsigned r = v.u + 0x7FFFu + ((v.u >> 16) & 1u);
  union { unsigned short s; bf16 b; } o; o.s = (unsigned short)(r >> 16);
  return o.b;
}
static __device__ __forceinline__ float bf2f(bf16 b) {
  union { unsigned short s; bf16 b; } i; i.b = b;
  union { unsigned u; float f; } o; o.u = ((unsigned)i.s) << 16;
  return o.f;
}

// ---------- WMMA fragment load ----------
// 16x32 bf16 operand in the CDNA5 16-bit A layout (also used for B when each
// matrix column's K values are contiguous with stride ld between columns):
//   lane&15  -> row (or column)
//   lane>>4  -> K chunk: K[8h..8h+7] in VGPR0-3, K[8h+16..8h+23] in VGPR4-7
static __device__ __forceinline__ v16bf ld_frag(const bf16* p, int ld) {
  const int lane = threadIdx.x & 31;
  const int row  = lane & 15;
  const int koff = (lane >> 4) << 3;
  const bf16* base = p + row * ld + koff;
  union { v16bf v; u32x4 q[2]; } f;
  f.q[0] = *(const u32x4*)(base);
  f.q[1] = *(const u32x4*)(base + 16);
  return f.v;
}

static __device__ __forceinline__ v8f wmma_bf16(v16bf a, v16bf b, v8f c) {
  return __builtin_amdgcn_wmma_f32_16x16x32_bf16(false, a, false, b,
                                                 (short)0, c, false, false);
}

// Problem constants: B=16, C=256, H=W=64, N=4096, O=192, dq=32, dv=128, M=1024
#define NB   16
#define NC   256
#define NPIX 4096
#define NO   192
#define NDQ  32
#define NDV  128
#define NM   1024

// ---------- stage 0: weights -> bf16 ----------
__global__ void k_prep_weights(const float* __restrict__ wqkv,
                               const float* __restrict__ wout,
                               bf16* __restrict__ wqkv_bf,
                               bf16* __restrict__ wout_bf) {
  int i = blockIdx.x * blockDim.x + threadIdx.x;
  if (i < NO * NC)  wqkv_bf[i] = f2bf(wqkv[i]);
  if (i < NC * NDV) wout_bf[i] = f2bf(wout[i]);
}

// ---------- stage 1: imgs[b][c][n] -> bf16 imgsT[b][n][c], LDS-tiled ----------
__global__ void k_imgsT(const float* __restrict__ imgs, bf16* __restrict__ imgsT) {
  __shared__ float tile[32][33];
  const int n0 = blockIdx.x * 32;
  const int c0 = blockIdx.y * 32;
  const int b  = blockIdx.z;
  const int tx = threadIdx.x, ty = threadIdx.y;   // (32,8)
  const float* src = imgs + (((size_t)b * NC + c0) << 12) + n0;
  #pragma unroll
  for (int j = 0; j < 4; ++j)
    tile[ty + 8 * j][tx] = src[((size_t)(ty + 8 * j) << 12) + tx];
  __syncthreads();
  #pragma unroll
  for (int j = 0; j < 4; ++j)
    imgsT[(((size_t)b * NPIX + n0 + ty + 8 * j) << 8) + c0 + tx] =
        f2bf(tile[tx][ty + 8 * j]);
}

// ---------- stage 2: QKV GEMM with B-fragment reuse across all 12 o-tiles ----
__global__ void k_qkv_gemm(const bf16* __restrict__ wqkv_bf,
                           const bf16* __restrict__ imgsT,
                           bf16* __restrict__ qkv_bf) {
  const int wave = threadIdx.x >> 5;                  // 4 waves / block
  const int n0 = (blockIdx.x * 4 + wave) * 16;        // pixel tile
  const int b  = blockIdx.y;
  const bf16* Bm = imgsT + (size_t)b * NPIX * NC + (size_t)n0 * NC;
  const v8f vzero = {0.f,0.f,0.f,0.f,0.f,0.f,0.f,0.f};
  v16bf bfr[8];
  #pragma unroll
  for (int k = 0; k < 8; ++k) bfr[k] = ld_frag(Bm + k * 32, NC);
  const int lane = threadIdx.x & 31;
  const int col = lane & 15, rhi = (lane >> 4) << 3;
  for (int ot = 0; ot < NO / 16; ++ot) {
    const bf16* A = wqkv_bf + ot * 16 * NC;
    v8f acc = vzero;
    #pragma unroll
    for (int k = 0; k < 8; ++k)
      acc = wmma_bf16(ld_frag(A + k * 32, NC), bfr[k], acc);
    #pragma unroll
    for (int r = 0; r < 8; ++r) {
      int o = ot * 16 + r + rhi;
      qkv_bf[(((size_t)b * NO + o) << 12) + n0 + col] = f2bf(acc[r]);
    }
  }
}

// ---------- stage 3: pack q (tiled transpose) / maxpool k,v ----------
__global__ void k_pack_q(const bf16* __restrict__ qkv_bf, bf16* __restrict__ qT) {
  __shared__ bf16 tile[32][33];
  const int n0 = blockIdx.x * 32;
  const int b  = blockIdx.y;
  const int tx = threadIdx.x, ty = threadIdx.y;   // (32,8)
  const bf16* src = qkv_bf + (((size_t)b * NO) << 12) + n0;  // q = channels 0..31
  #pragma unroll
  for (int j = 0; j < 4; ++j)
    tile[ty + 8 * j][tx] = src[((size_t)(ty + 8 * j) << 12) + tx];
  __syncthreads();
  bf16* dst = qT + ((size_t)b * NPIX + n0) * NDQ;
  #pragma unroll
  for (int j = 0; j < 4; ++j)
    dst[(size_t)(ty + 8 * j) * NDQ + tx] = tile[tx][ty + 8 * j];
}
__global__ void k_pack_k(const bf16* __restrict__ qkv_bf, bf16* __restrict__ kT) {
  int i = blockIdx.x * blockDim.x + threadIdx.x;     // [0, 16*1024*32)
  int d = i & 31, m = (i >> 5) & 1023, b = i >> 15;
  int hh = m >> 5, ww = m & 31;
  const bf16* src = qkv_bf + (((size_t)b * NO + NDQ + d) << 12) + (hh * 2) * 64 + ww * 2;
  float v = fmaxf(fmaxf(bf2f(src[0]), bf2f(src[1])),
                  fmaxf(bf2f(src[64]), bf2f(src[65])));
  kT[i] = f2bf(v);                                   // kT[b][m][d]
}
__global__ void k_pack_v(const bf16* __restrict__ qkv_bf, bf16* __restrict__ vM) {
  int i = blockIdx.x * blockDim.x + threadIdx.x;     // [0, 16*128*1024)
  int m = i & 1023, d = (i >> 10) & 127, b = i >> 17;
  int hh = m >> 5, ww = m & 31;
  const bf16* src = qkv_bf + (((size_t)b * NO + 2 * NDQ + d) << 12) + (hh * 2) * 64 + ww * 2;
  float v = fmaxf(fmaxf(bf2f(src[0]), bf2f(src[1])),
                  fmaxf(bf2f(src[64]), bf2f(src[65])));
  vM[i] = f2bf(v);                                   // vM[b][d][m], m contiguous
}

// ---------- stage 4: flash attention ----------
// 4 waves / block, one 16-query tile per wave.  K/V m-chunk tiles are staged
// cooperatively into *double-buffered* LDS with gfx1250 async loads
// (ASYNCcnt-tracked; in-order completion lets s_wait_asynccnt 5 mean "current
// tile done" while the next tile's 5 ops stay in flight).  Output drains
// through LDS via global_store_async_from_lds_b128 for coalesced 16B stores.
__global__ void k_attn(const bf16* __restrict__ qT, const bf16* __restrict__ kT,
                       const bf16* __restrict__ vM, bf16* __restrict__ outT) {
  const int wave = threadIdx.x >> 5;
  const int lane = threadIdx.x & 31;
  const int n0 = (blockIdx.x * 4 + wave) * 16;
  const int b  = blockIdx.y;
  __shared__ bf16 ktile[2][32 * 32];              // [m:32][d:32]   2x2 KB
  __shared__ bf16 vtile[2][NDV * 32];             // [d:128][m:32]  2x8 KB
  __shared__ bf16 pbuf[4][16 * 32];               // per-wave P staging
  __shared__ bf16 obuf[4][16 * NDV];              // per-wave O staging
  bf16* pw = pbuf[wave];
  const int col = lane & 15, rhi = (lane >> 4) << 3;

  const v16bf a_q = ld_frag(qT + ((size_t)b * NPIX + n0) * NDQ, NDQ);
  const bf16* kBp = kT + (size_t)b * NM * NDQ;    // block-uniform -> SGPR pair
  const bf16* vBp = vM + (size_t)b * NDV * NM;

  const unsigned kt_base = (unsigned)(uintptr_t)&ktile[0][0];  // LDS byte offs
  const unsigned vt_base = (unsigned)(uintptr_t)&vtile[0][0];
  const int tid  = (int)threadIdx.x;
  const int trow = tid >> 2;                      // 0..31  (16B chunk layout)
  const int tchk = (tid & 3) * 16;

  // issue the 5 async loads (1 K + 4 V) for one 32-m tile into buffer `sel`
  auto issue_tile = [&](int m0, int sel) {
    unsigned klds = kt_base + (unsigned)sel * (32 * 32 * 2) + trow * 64 + tchk;
    unsigned kgo  = (unsigned)((m0 + trow) * (NDQ * 2)) + tchk;
    asm volatile("global_load_async_to_lds_b128 %0, %1, %2 offset:0"
                 :: "v"(klds), "v"(kgo), "s"(kBp) : "memory");
    #pragma unroll
    for (int i = 0; i < 4; ++i) {
      int t = i * 128 + tid;
      unsigned vlds = vt_base + (unsigned)sel * (NDV * 32 * 2)
                      + (t >> 2) * 64 + (t & 3) * 16;
      unsigned vgo  = (unsigned)((t >> 2) * (NM * 2) + m0 * 2 + (t & 3) * 16);
      asm volatile("global_load_async_to_lds_b128 %0, %1, %2 offset:0"
                   :: "v"(vlds), "v"(vgo), "s"(vBp) : "memory");
    }
  };

  const v8f vzero = {0.f,0.f,0.f,0.f,0.f,0.f,0.f,0.f};
  float mrow[8], lrow[8];
  v8f o_acc[8];
  #pragma unroll
  for (int r = 0; r < 8; ++r) { mrow[r] = -3.0e38f; lrow[r] = 0.f; }
  #pragma unroll
  for (int dc = 0; dc < 8; ++dc) o_acc[dc] = vzero;

  issue_tile(0, 0);                               // prologue: prime both buffers
  issue_tile(32, 1);

  for (int it = 0; it < NM / 32; ++it) {
    // current tile's 5 ops complete (in-order) -> <=5 outstanding if the next
    // tile is in flight, ==0 on the final iteration
    if (it + 1 < NM / 32) asm volatile("s_wait_asynccnt 5" ::: "memory");
    else                  asm volatile("s_wait_asynccnt 0" ::: "memory");
    __syncthreads();                              // tile visible to all waves

    const bf16* kt = ktile[it & 1];
    const bf16* vt = vtile[it & 1];
    v8f s1 = wmma_bf16(a_q, ld_frag(kt, 32), vzero);            // m0..m0+15
    v8f s2 = wmma_bf16(a_q, ld_frag(kt + 16 * 32, 32), vzero);  // +16..+31
    #pragma unroll
    for (int r = 0; r < 8; ++r) {
      float rmax = fmaxf(s1[r], s2[r]);
      #pragma unroll
      for (int off = 8; off; off >>= 1)
        rmax = fmaxf(rmax, __shfl_xor(rmax, off, 32));
      float mnew = fmaxf(mrow[r], rmax);
      float corr = __expf(mrow[r] - mnew);
      float p1 = __expf(s1[r] - mnew);
      float p2 = __expf(s2[r] - mnew);
      float rsum = p1 + p2;
      #pragma unroll
      for (int off = 8; off; off >>= 1)
        rsum += __shfl_xor(rsum, off, 32);
      lrow[r] = lrow[r] * corr + rsum;
      mrow[r] = mnew;
      #pragma unroll
      for (int dc = 0; dc < 8; ++dc) o_acc[dc][r] *= corr;
      pw[(r + rhi) * 32 + col]      = f2bf(p1);
      pw[(r + rhi) * 32 + 16 + col] = f2bf(p2);
    }
    asm volatile("s_wait_dscnt 0" ::: "memory");  // wave-private P staging
    const v16bf a_p = ld_frag(pw, 32);
    #pragma unroll
    for (int dc = 0; dc < 8; ++dc)
      o_acc[dc] = wmma_bf16(a_p, ld_frag(vt + (dc * 16) * 32, 32), o_acc[dc]);

    __syncthreads();                              // all waves done with tile it
    if (it + 2 < NM / 32) issue_tile((it + 2) * 32, it & 1);
  }

  // Epilogue: scatter O into wave-private LDS, then drain with coalesced
  // async LDS->global 16B stores.
  bf16* ow = obuf[wave];
  #pragma unroll
  for (int dc = 0; dc < 8; ++dc) {
    #pragma unroll
    for (int r = 0; r < 8; ++r)
      ow[(r + rhi) * NDV + dc * 16 + col] = f2bf(o_acc[dc][r] / lrow[r]);
  }
  asm volatile("s_wait_dscnt 0" ::: "memory");
  const unsigned ob_base = (unsigned)(uintptr_t)ow;
  const bf16* oBp = outT + ((size_t)b * NPIX + n0) * NDV;  // wave-uniform
  #pragma unroll
  for (int j = 0; j < 8; ++j) {                   // 16 rows x 16 chunks of 16B
    int t2 = j * 32 + lane;
    int row = t2 >> 4, chk = t2 & 15;
    const bf16* gaddr = oBp + row * NDV + chk * 8;
    unsigned lsrc = ob_base + row * (NDV * 2) + chk * 16;
    asm volatile("global_store_async_from_lds_b128 %0, %1, off"
                 :: "v"(gaddr), "v"(lsrc) : "memory");
  }
  asm volatile("s_wait_asynccnt 0" ::: "memory");
}

// ---------- stage 5: out = imgs + scale * (w_out x attn_out) ----------
// B fragments (outT) kept in registers, reused across all 16 c-tiles.
__global__ void k_out_proj(const bf16* __restrict__ wout_bf,
                           const bf16* __restrict__ outT,
                           const float* __restrict__ imgs,
                           const float* __restrict__ scale_p,
                           float* __restrict__ out) {
  const int wave = threadIdx.x >> 5;                  // 4 waves / block
  const int n0 = (blockIdx.x * 4 + wave) * 16;
  const int b  = blockIdx.y;
  const bf16* Bm = outT + (((size_t)b * NPIX + n0) << 7);
  const v8f vzero = {0.f,0.f,0.f,0.f,0.f,0.f,0.f,0.f};
  v16bf bfr[4];
  #pragma unroll
  for (int k = 0; k < 4; ++k) bfr[k] = ld_frag(Bm + k * 32, NDV);
  const float s = *scale_p;
  const int lane = threadIdx.x & 31;
  const int col = lane & 15, rhi = (lane >> 4) << 3;
  for (int ct = 0; ct < NC / 16; ++ct) {
    const bf16* A = wout_bf + ct * 16 * NDV;
    v8f acc = vzero;
    #pragma unroll
    for (int k = 0; k < 4; ++k)
      acc = wmma_bf16(ld_frag(A + k * 32, NDV), bfr[k], acc);
    #pragma unroll
    for (int r = 0; r < 8; ++r) {
      int c = ct * 16 + r + rhi;
      size_t idx = (((size_t)b * NC + c) << 12) + n0 + col;
      out[idx] = imgs[idx] + s * acc[r];
    }
  }
}

extern "C" void kernel_launch(void* const* d_in, const int* in_sizes, int n_in,
                              void* d_out, int out_size, void* d_ws, size_t ws_size,
                              hipStream_t stream) {
  (void)in_sizes; (void)n_in; (void)out_size; (void)ws_size;
  const float* imgs  = (const float*)d_in[0];
  const float* wqkv  = (const float*)d_in[1];
  const float* wout  = (const float*)d_in[2];
  const float* scale = (const float*)d_in[3];
  float* out = (float*)d_out;

  char* ws = (char*)d_ws;
  size_t off = 0;
  auto walloc = [&](size_t bytes) {
    char* p = ws + off;
    off += (bytes + 255) & ~(size_t)255;
    return p;
  };
  bf16* wqkv_bf = (bf16*)walloc((size_t)NO * NC * 2);
  bf16* wout_bf = (bf16*)walloc((size_t)NC * NDV * 2);
  bf16* imgsT   = (bf16*)walloc((size_t)NB * NPIX * NC * 2);   // 32 MB
  bf16* qkv_bf  = (bf16*)walloc((size_t)NB * NO * NPIX * 2);   // 24 MB
  bf16* qT      = (bf16*)walloc((size_t)NB * NPIX * NDQ * 2);  // 4 MB
  bf16* kT      = (bf16*)walloc((size_t)NB * NM * NDQ * 2);    // 1 MB
  bf16* vM      = (bf16*)walloc((size_t)NB * NDV * NM * 2);    // 4 MB
  bf16* outT    = (bf16*)walloc((size_t)NB * NPIX * NDV * 2);  // 16 MB

  k_prep_weights<<<(NO * NC + 255) / 256, 256, 0, stream>>>(wqkv, wout, wqkv_bf, wout_bf);

  dim3 gT(NPIX / 32, NC / 32, NB);
  k_imgsT<<<gT, dim3(32, 8), 0, stream>>>(imgs, imgsT);

  dim3 g_qkv(NPIX / 64, NB);                   // 4 waves/block, B frags reused
  k_qkv_gemm<<<g_qkv, 128, 0, stream>>>(wqkv_bf, imgsT, qkv_bf);

  dim3 gQ(NPIX / 32, NB);
  k_pack_q<<<gQ, dim3(32, 8), 0, stream>>>(qkv_bf, qT);
  k_pack_k<<<(NB * NM * NDQ) / 256, 256, 0, stream>>>(qkv_bf, kT);
  k_pack_v<<<(NB * NDV * NM) / 256, 256, 0, stream>>>(qkv_bf, vM);

  dim3 g_attn(NPIX / 64, NB);                  // 4 waves/block, shared K/V tiles
  k_attn<<<g_attn, 128, 0, stream>>>(qT, kT, vM, outT);

  dim3 g_out(NPIX / 64, NB);                   // 4 waves/block, B frags reused
  k_out_proj<<<g_out, 128, 0, stream>>>(wout_bf, outT, imgs, scale, out);
}
